// SelfAttention_41386304864294
// MI455X (gfx1250) — matrix operand order repaired and checked by reading, not compiled, not run on hardware
//
#include <hip/hip_runtime.h>
#include <hip/hip_bf16.h>
#include <stdint.h>

// ---------------------------------------------------------------------------
// CDNA5 (gfx1250) wave32 WMMA self-attention, software-pipelined K loops.
//   D = A(16x32 bf16) x B(32x16 bf16) + C(16x16 f32)  via
//   __builtin_amdgcn_wmma_f32_16x16x32_bf16
// A/B fragments use contiguous K order (k = 16*kg + s), not the ISA table order.
// ---------------------------------------------------------------------------

typedef __attribute__((ext_vector_type(16))) __bf16 bf16x16;
typedef __attribute__((ext_vector_type(8)))  float  f32x8;

union FragAB {
  bf16x16      v;
  unsigned int u[8];
};

__device__ __forceinline__ unsigned short f32_to_bf16_rne(float f) {
  unsigned int u = __float_as_uint(f);
  u += 0x7FFFu + ((u >> 16) & 1u);      // round-to-nearest-even
  return (unsigned short)(u >> 16);
}

// A-matrix (16x32, 16-bit) per-VGPR K offset:
//   lanes 0-15 (kg=0): v0..v7 -> K {0..15}
//   lanes 16-31 (kg=1): v0..v7 -> K {16..31}
__device__ __forceinline__ int koffA(int j, int kg) {
  return 16 * kg + 2 * j;
}

__device__ __forceinline__ f32x8 wmma_bf16(const FragAB& a, const FragAB& b, f32x8 c) {
  return __builtin_amdgcn_wmma_f32_16x16x32_bf16(false, a.v, false, b.v,
                                                 (short)0, c, false, false);
}

__device__ __forceinline__ void load_fragA(FragAB& a, const unsigned short* arow,
                                           int k0, int kg) {
#pragma unroll
  for (int j = 0; j < 8; ++j)
    a.u[j] = *(const unsigned int*)&arow[k0 + koffA(j, kg)];
}

// B fragment: lane column fixed; contiguous K pairs along a [N][K] row
__device__ __forceinline__ void load_fragB(FragAB& b, const unsigned short* bcol) {
#pragma unroll
  for (int j = 0; j < 8; ++j)
    b.u[j] = *(const unsigned int*)&bcol[2 * j];
}

// ---------------------------------------------------------------------------
// Phase 0: precision conversion / weight transpose
// ---------------------------------------------------------------------------
__global__ void cvt_f32_bf16(const float* __restrict__ in,
                             unsigned short* __restrict__ out, int n) {
  int i = blockIdx.x * blockDim.x + threadIdx.x;
  if (i < n) out[i] = f32_to_bf16_rne(in[i]);
}

// in: [K][N] f32 row-major  ->  out: [N][K] bf16 row-major
__global__ void transpose_f32_bf16(const float* __restrict__ in,
                                   unsigned short* __restrict__ out,
                                   int K, int N) {
  int i = blockIdx.x * blockDim.x + threadIdx.x;
  if (i < K * N) {
    int k = i / N, n = i % N;
    out[(size_t)n * K + k] = f32_to_bf16_rne(in[i]);
  }
}

// ---------------------------------------------------------------------------
// Pipelined 16x64-per-wave GEMM core: acc[4] += A[16xK] * Bt[64 cols][K]
// Ping-pong fragment registers, unroll-by-2 over K (K % 64 == 0).
// ---------------------------------------------------------------------------
__device__ __forceinline__ void gemm_core_16x64(
    const unsigned short* __restrict__ arow,   // A row base (lane-specific)
    const unsigned short* __restrict__ bcol0,  // Bt col base +16*kg (lane-specific)
    int K, int kg, f32x8 acc[4]) {
  FragAB a0, a1, b0[4], b1[4];
  load_fragA(a0, arow, 0, kg);
#pragma unroll
  for (int nt = 0; nt < 4; ++nt) load_fragB(b0[nt], bcol0 + (size_t)nt * 16 * K);

  for (int k0 = 0; k0 < K; k0 += 64) {
    // issue loads for k0+32 before computing on k0
    load_fragA(a1, arow, k0 + 32, kg);
#pragma unroll
    for (int nt = 0; nt < 4; ++nt)
      load_fragB(b1[nt], bcol0 + (size_t)nt * 16 * K + k0 + 32);
#pragma unroll
    for (int nt = 0; nt < 4; ++nt) acc[nt] = wmma_bf16(a0, b0[nt], acc[nt]);

    if (k0 + 64 < K) {
      __builtin_prefetch(&arow[k0 + 128], 0, 1);
      load_fragA(a0, arow, k0 + 64, kg);
#pragma unroll
      for (int nt = 0; nt < 4; ++nt)
        load_fragB(b0[nt], bcol0 + (size_t)nt * 16 * K + k0 + 64);
    }
#pragma unroll
    for (int nt = 0; nt < 4; ++nt) acc[nt] = wmma_bf16(a1, b1[nt], acc[nt]);
  }
}

// ---------------------------------------------------------------------------
// Phase 1: qkv = x @ W_attn  (bf16 WMMA GEMM, scatter epilogue)
//   Q,K -> [b*H+h][t][64] bf16 ; V -> [b*H+h][64][T] bf16 (transposed)
// ---------------------------------------------------------------------------
__global__ void __launch_bounds__(256)
gemm_qkv(const unsigned short* __restrict__ A,
         const unsigned short* __restrict__ Bt,
         unsigned short* __restrict__ Qb,
         unsigned short* __restrict__ Kb,
         unsigned short* __restrict__ Vt,
         int M, int N, int K, int T, int H, int D, int C) {
  const int lane = threadIdx.x & 31;
  const int wid  = (blockIdx.x * blockDim.x + threadIdx.x) >> 5;
  const int tilesN = N >> 6;
  if (wid >= (M >> 4) * tilesN) return;      // wave-uniform (EXEC all-1s for WMMA)
  const int tn = wid % tilesN;
  const int tm = wid / tilesN;
  const int m  = lane & 15;
  const int kg = lane >> 4;

  const unsigned short* arow  = A + (size_t)(tm * 16 + m) * K;
  const unsigned short* bcol0 = Bt + (size_t)(tn * 64 + m) * K + 16 * kg;

  f32x8 acc[4];
  const f32x8 fzero = {0.f, 0.f, 0.f, 0.f, 0.f, 0.f, 0.f, 0.f};
#pragma unroll
  for (int nt = 0; nt < 4; ++nt) acc[nt] = fzero;

  gemm_core_16x64(arow, bcol0, K, kg, acc);

  // scatter epilogue: split 3C columns into Q / K / V(T-transposed)
#pragma unroll
  for (int nt = 0; nt < 4; ++nt) {
#pragma unroll
    for (int j = 0; j < 8; ++j) {
      const int rg = tm * 16 + ((lane & 16) ? j + 8 : j);  // global row = b*T + t
      const int cg = tn * 64 + nt * 16 + m;                // global col in [0, 3C)
      const int bb = rg / T, t = rg % T;
      const int sect = (cg >= 2 * C) ? 2 : ((cg >= C) ? 1 : 0);
      const int c = cg - sect * C;
      const int h = c / D, d = c % D;
      const unsigned short val = f32_to_bf16_rne(acc[nt][j]);
      const size_t head = (size_t)(bb * H + h);
      if (sect == 0)      Qb[(head * T + t) * D + d] = val;
      else if (sect == 1) Kb[(head * T + t) * D + d] = val;
      else                Vt[(head * D + d) * T + t] = val;
    }
  }
}

// ---------------------------------------------------------------------------
// Phase 2: causal flash attention, one wave per 16 query rows.
//   S-tile 16x32 = 4 WMMAs; V fragments loaded BEFORE the softmax so global
//   latency overlaps the VALU softmax + LDS P re-layout; then 4 PV WMMAs.
// ---------------------------------------------------------------------------
__global__ void __launch_bounds__(256)
attn_fwd(const unsigned short* __restrict__ Qb,
         const unsigned short* __restrict__ Kb,
         const unsigned short* __restrict__ Vt,
         unsigned short* __restrict__ Y,
         int B, int T, int H, int D, int C, float scale) {
  __shared__ unsigned short pshm[8 * 16 * 32];  // per-wave 16x32 bf16 scratch
  const int lane = threadIdx.x & 31;
  const int slot = threadIdx.x >> 5;
  const int wid  = (blockIdx.x * blockDim.x + threadIdx.x) >> 5;
  const int qtiles = T >> 4;
  if (wid >= B * H * qtiles) return;
  const int qt = wid % qtiles;
  const int bh = wid / qtiles;
  const int bb = bh / H, h = bh % H;
  const int q0 = qt << 4;
  const int m  = lane & 15;
  const int kg = lane >> 4;

  const unsigned short* Qh = Qb + (size_t)bh * T * D;
  const unsigned short* Kh = Kb + (size_t)bh * T * D;
  const unsigned short* Vh = Vt + (size_t)bh * D * T;
  unsigned short* pbase = &pshm[slot * 512];

  // Q fragments: two 16x32 chunks of D=64
  FragAB aq[2];
  {
    const unsigned short* qrow = Qh + (size_t)(q0 + m) * D;
#pragma unroll
    for (int f = 0; f < 2; ++f)
      load_fragA(aq[f], qrow, 32 * f, kg);
  }

  f32x8 yacc[4];
  const f32x8 fzero = {0.f, 0.f, 0.f, 0.f, 0.f, 0.f, 0.f, 0.f};
#pragma unroll
  for (int nt = 0; nt < 4; ++nt) yacc[nt] = fzero;
  float mrow[8], lrow[8];
#pragma unroll
  for (int j = 0; j < 8; ++j) { mrow[j] = -3.0e38f; lrow[j] = 0.f; }

  for (int j0 = 0; j0 < q0 + 16; j0 += 32) {
    // ---- S = Q K^T for 32 keys (two 16-column halves) ----
    f32x8 sl = fzero, sr = fzero;
    const unsigned short* klp = Kh + (size_t)(j0 + m) * D + 16 * kg;
    const unsigned short* krp = Kh + (size_t)(j0 + 16 + m) * D + 16 * kg;
#pragma unroll
    for (int f = 0; f < 2; ++f) {
      FragAB bkl, bkr;
      load_fragB(bkl, klp + 32 * f);
      load_fragB(bkr, krp + 32 * f);
      sl = wmma_bf16(aq[f], bkl, sl);
      sr = wmma_bf16(aq[f], bkr, sr);
    }

    // ---- issue V fragment loads now; they complete during the softmax ----
    FragAB bv[4];
#pragma unroll
    for (int nt = 0; nt < 4; ++nt)
      load_fragB(bv[nt], Vh + (size_t)(nt * 16 + m) * T + j0 + 16 * kg);

    // ---- online softmax over the 32-key block ----
    float alpha[8];
#pragma unroll
    for (int j = 0; j < 8; ++j) {
      const int qr  = q0 + ((lane & 16) ? j + 8 : j);
      const int kcl = j0 + m;
      float vl = sl[j] * scale; if (kcl > qr)      vl = -3.0e38f;   // causal mask
      float vr = sr[j] * scale; if (kcl + 16 > qr) vr = -3.0e38f;
      float mx = fmaxf(vl, vr);
      mx = fmaxf(mx, __shfl_xor(mx, 1, 32));
      mx = fmaxf(mx, __shfl_xor(mx, 2, 32));
      mx = fmaxf(mx, __shfl_xor(mx, 4, 32));
      mx = fmaxf(mx, __shfl_xor(mx, 8, 32));
      const float mnew = fmaxf(mrow[j], mx);
      const float a  = __expf(mrow[j] - mnew);
      const float pl = __expf(vl - mnew);
      const float pr = __expf(vr - mnew);
      float rs = pl + pr;
      rs += __shfl_xor(rs, 1, 32);
      rs += __shfl_xor(rs, 2, 32);
      rs += __shfl_xor(rs, 4, 32);
      rs += __shfl_xor(rs, 8, 32);
      lrow[j] = lrow[j] * a + rs;
      mrow[j] = mnew;
      alpha[j] = a;
      // stash P (C-layout -> LDS row-major 16x32 bf16)
      const int prow = (lane & 16) ? j + 8 : j;
      pbase[prow * 32 + m]      = f32_to_bf16_rne(pl);
      pbase[prow * 32 + m + 16] = f32_to_bf16_rne(pr);
    }
#pragma unroll
    for (int nt = 0; nt < 4; ++nt)
#pragma unroll
      for (int j = 0; j < 8; ++j) yacc[nt][j] *= alpha[j];

    // ---- reload P as a 16x32 A-fragment (compiler orders ds store->load) ----
    FragAB pa;
#pragma unroll
    for (int j = 0; j < 8; ++j)
      pa.u[j] = *(const unsigned int*)&pbase[m * 32 + koffA(j, kg)];

    // ---- Y += P x V  (V fragments already resident) ----
#pragma unroll
    for (int nt = 0; nt < 4; ++nt)
      yacc[nt] = wmma_bf16(pa, bv[nt], yacc[nt]);
  }

  // ---- normalize and store y as bf16 [b][t][h*D + d] ----
#pragma unroll
  for (int j = 0; j < 8; ++j) {
    const float inv = 1.0f / lrow[j];
    const int row = q0 + ((lane & 16) ? j + 8 : j);
    unsigned short* yrow = Y + ((size_t)(bb * T + row)) * C + h * D;
#pragma unroll
    for (int nt = 0; nt < 4; ++nt)
      yrow[nt * 16 + m] = f32_to_bf16_rne(yacc[nt][j] * inv);
  }
}

// ---------------------------------------------------------------------------
// Phase 3: out = y @ W_proj (bf16 WMMA GEMM, f32 output)
// ---------------------------------------------------------------------------
__global__ void __launch_bounds__(256)
gemm_proj(const unsigned short* __restrict__ A,
          const unsigned short* __restrict__ Bt,
          float* __restrict__ out, int M, int N, int K) {
  const int lane = threadIdx.x & 31;
  const int wid  = (blockIdx.x * blockDim.x + threadIdx.x) >> 5;
  const int tilesN = N >> 6;
  if (wid >= (M >> 4) * tilesN) return;
  const int tn = wid % tilesN;
  const int tm = wid / tilesN;
  const int m  = lane & 15;
  const int kg = lane >> 4;

  const unsigned short* arow  = A + (size_t)(tm * 16 + m) * K;
  const unsigned short* bcol0 = Bt + (size_t)(tn * 64 + m) * K + 16 * kg;

  f32x8 acc[4];
  const f32x8 fzero = {0.f, 0.f, 0.f, 0.f, 0.f, 0.f, 0.f, 0.f};
#pragma unroll
  for (int nt = 0; nt < 4; ++nt) acc[nt] = fzero;

  gemm_core_16x64(arow, bcol0, K, kg, acc);

#pragma unroll
  for (int nt = 0; nt < 4; ++nt) {
#pragma unroll
    for (int j = 0; j < 8; ++j) {
      const int rg = tm * 16 + ((lane & 16) ? j + 8 : j);
      const int cg = tn * 64 + nt * 16 + m;
      out[(size_t)rg * N + cg] = acc[nt][j];
    }
  }
}

// ---------------------------------------------------------------------------
// Host launch
// ---------------------------------------------------------------------------
extern "C" void kernel_launch(void* const* d_in, const int* in_sizes, int n_in,
                              void* d_out, int out_size, void* d_ws, size_t ws_size,
                              hipStream_t stream) {
  (void)in_sizes; (void)n_in; (void)out_size; (void)ws_size;
  constexpr int B = 4, T = 2048, C = 1024, H = 16, D = 64;
  constexpr int M  = B * T;    // 8192
  constexpr int N3 = 3 * C;    // 3072

  const float* x      = (const float*)d_in[0];
  const float* W_attn = (const float*)d_in[1];
  const float* W_proj = (const float*)d_in[2];
  float* out = (float*)d_out;

  uint8_t* ws = (uint8_t*)d_ws;
  size_t off = 0;
  auto walloc = [&](size_t bytes) -> unsigned short* {
    unsigned short* p = (unsigned short*)(ws + off);
    off += (bytes + 255) & ~(size_t)255;
    return p;
  };
  // total workspace: ~92 MB of bf16 staging
  unsigned short* xb  = walloc((size_t)M * C * 2);    // x bf16 [M][C]
  unsigned short* Wat = walloc((size_t)N3 * C * 2);   // W_attn^T bf16 [3C][C]
  unsigned short* Wpt = walloc((size_t)C * C * 2);    // W_proj^T bf16 [C][C]
  unsigned short* Qb  = walloc((size_t)M * C * 2);    // [b*H+h][t][64]
  unsigned short* Kb  = walloc((size_t)M * C * 2);    // [b*H+h][t][64]
  unsigned short* Vt  = walloc((size_t)M * C * 2);    // [b*H+h][64][T]
  unsigned short* Yb  = walloc((size_t)M * C * 2);    // y bf16 [M][C]

  { int n = M * C;  cvt_f32_bf16<<<(n + 255) / 256, 256, 0, stream>>>(x, xb, n); }
  { int n = C * N3; transpose_f32_bf16<<<(n + 255) / 256, 256, 0, stream>>>(W_attn, Wat, C, N3); }
  { int n = C * C;  transpose_f32_bf16<<<(n + 255) / 256, 256, 0, stream>>>(W_proj, Wpt, C, C); }

  { // QKV GEMM: (M/16)*(N3/64) waves, 8 waves per block
    int waves = (M / 16) * (N3 / 64);
    gemm_qkv<<<waves / 8, 256, 0, stream>>>(xb, Wat, Qb, Kb, Vt, M, N3, C, T, H, D, C);
  }
  { // attention: one wave per 16 query rows per head
    int waves = B * H * (T / 16);
    attn_fwd<<<waves / 8, 256, 0, stream>>>(Qb, Kb, Vt, Yb, B, T, H, D, C, 0.125f);
  }
  { // output projection
    int waves = (M / 16) * (C / 64);
    gemm_proj<<<waves / 8, 256, 0, stream>>>(Yb, Wpt, out, M, C, C);
  }
}
